// GAT_40295383171398
// MI455X (gfx1250) — matrix operand order repaired and checked by reading, not compiled, not run on hardware
//
#include <hip/hip_runtime.h>
#include <hip/hip_bf16.h>
#include <math.h>

typedef __attribute__((ext_vector_type(16))) _Float16 v16h;
typedef __attribute__((ext_vector_type(8)))  _Float16 v8h;
typedef __attribute__((ext_vector_type(8)))  float    v8f;

namespace {
constexpr int NN    = 30000;
constexpr int EE    = 480000;
constexpr int INC   = 256;
constexpr int HIDC  = 64;
constexpr int HEADSC= 8;
constexpr int OUTC  = 128;
constexpr int GC    = 512;
constexpr int HCC   = HEADSC * HIDC;   // 512
constexpr int ETOT  = EE + NN;         // 510000 (self loops appended)
}

#define CDIV(a,b) (((a)+(b)-1)/(b))

// ---------------- elementwise / utility kernels ----------------

__global__ void k_fill(float* p, float v, int n) {
  int i = blockIdx.x * blockDim.x + threadIdx.x;
  if (i < n) p[i] = v;
}

__global__ void k_f2h(const float* __restrict__ s, _Float16* __restrict__ d, int n) {
  int i = blockIdx.x * blockDim.x + threadIdx.x;
  if (i < n) d[i] = (_Float16)s[i];
}

// W [K, Nc] f32 row-major  ->  Wt [Nc, K] f16 (transposed for B-fragment loads)
__global__ void k_transpose_f2h(const float* __restrict__ W, _Float16* __restrict__ Wt,
                                int K, int Nc) {
  int i = blockIdx.x * blockDim.x + threadIdx.x;
  if (i < K * Nc) {
    int k = i / Nc, n = i % Nc;
    Wt[(size_t)n * K + k] = (_Float16)W[i];
  }
}

// ---------------- WMMA GEMM: C[M,Nc] = A[M,K] * B[K,Nc] ----------------
// A: f16 row-major. Bt: f16 [Nc,K] (B transposed). C: f32 row-major.
// block = 128 threads (4 waves); wave w computes cols [blockIdx.y*64 + w*16, +16)
// for rows [blockIdx.x*16, +16). K multiple of 32.
__global__ void k_gemm_wmma(const _Float16* __restrict__ A,
                            const _Float16* __restrict__ Bt,
                            float* __restrict__ C,
                            int M, int K, int Nc) {
  const int lane = threadIdx.x & 31;
  const int wave = threadIdx.x >> 5;
  const int hs   = lane >> 4;      // half-wave select
  const int l15  = lane & 15;
  const int row  = blockIdx.x * 16 + l15;           // A row held by this lane
  const int col  = blockIdx.y * 64 + wave * 16 + l15; // B/C column held by this lane
  const _Float16* __restrict__ Ar = A  + (size_t)row * K;
  const _Float16* __restrict__ Bc = Bt + (size_t)col * K;

  v8f acc = {};
  for (int kb = 0; kb < K; kb += 32) {
    // A fragment: lanes 0-15 -> K = kb+0..7, kb+16..23 ; lanes 16-31 -> K = kb+8..15, kb+24..31
    v8h a0 = *(const v8h*)(Ar + kb + hs * 8);
    v8h a1 = *(const v8h*)(Ar + kb + 16 + hs * 8);
    // B fragment: lanes 0-15 -> K = kb+0..7, kb+16..23 ; lanes 16-31 -> K = kb+8..15, kb+24..31 (col-contiguous in Bt)
    v8h b0 = *(const v8h*)(Bc + kb + hs * 8);
    v8h b1 = *(const v8h*)(Bc + kb + 16 + hs * 8);
    v16h av = __builtin_shufflevector(a0, a1, 0,1,2,3,4,5,6,7,8,9,10,11,12,13,14,15);
    v16h bv = __builtin_shufflevector(b0, b1, 0,1,2,3,4,5,6,7,8,9,10,11,12,13,14,15);
    acc = __builtin_amdgcn_wmma_f32_16x16x32_f16(
        /*neg_a=*/false, av, /*neg_b=*/false, bv,
        /*c_mod=*/(short)0, acc, /*reuse_a=*/false, /*reuse_b=*/false);
  }
  // D layout: lane holds column (lane&15); rows hs*8 .. hs*8+7 across the 8 floats
  float* Cp = C + (size_t)(blockIdx.x * 16 + hs * 8) * Nc + col;
  #pragma unroll
  for (int r = 0; r < 8; ++r) Cp[(size_t)r * Nc] = acc[r];
}

// ---------------- GAT edge/softmax kernels ----------------

// alpha_s[n,h] = <h[n,h,:], a_src[h,:]>, alpha_d likewise
__global__ void k_alpha(const float* __restrict__ h, const float* __restrict__ asrc,
                        const float* __restrict__ adst, float* __restrict__ als,
                        float* __restrict__ ald, int nNodes, int H, int C) {
  int i = blockIdx.x * blockDim.x + threadIdx.x;
  if (i >= nNodes * H) return;
  int n = i / H, hh = i % H;
  const float* hp = h + (size_t)n * H * C + (size_t)hh * C;
  const float* sp = asrc + (size_t)hh * C;
  const float* dp = adst + (size_t)hh * C;
  float vs = 0.f, vd = 0.f;
  for (int c = 0; c < C; ++c) { float x = hp[c]; vs += x * sp[c]; vd += x * dp[c]; }
  als[i] = vs; ald[i] = vd;
}

__device__ __forceinline__ void edge_sd(const int* ei, int e, int Eorig, int& s, int& d) {
  if (e < Eorig) { s = ei[e]; d = ei[Eorig + e]; }
  else           { s = e - Eorig; d = s; }          // appended self loops
}

__device__ __forceinline__ float leaky(float v) { return v > 0.f ? v : 0.2f * v; }

// monotone float max via integer atomics (works for mixed signs, init = -inf)
__device__ __forceinline__ void atomicMaxF32(float* addr, float v) {
  int iv = __float_as_int(v);
  if (iv >= 0) atomicMax((int*)addr, iv);
  else         atomicMin((unsigned int*)addr, (unsigned int)iv);
}

__global__ void k_edge_max(const int* __restrict__ ei, const float* __restrict__ als,
                           const float* __restrict__ ald, float* __restrict__ segmax,
                           int Etot, int Eorig, int H) {
  int i = blockIdx.x * blockDim.x + threadIdx.x;
  if (i >= Etot * H) return;
  int e = i / H, hh = i % H;
  int s, d; edge_sd(ei, e, Eorig, s, d);
  float v = leaky(als[(size_t)s * H + hh] + ald[(size_t)d * H + hh]);
  atomicMaxF32(&segmax[(size_t)d * H + hh], v);
}

__global__ void k_edge_exp(const int* __restrict__ ei, const float* __restrict__ als,
                           const float* __restrict__ ald, const float* __restrict__ segmax,
                           float* __restrict__ ew, float* __restrict__ segsum,
                           int Etot, int Eorig, int H) {
  int i = blockIdx.x * blockDim.x + threadIdx.x;
  if (i >= Etot * H) return;
  int e = i / H, hh = i % H;
  int s, d; edge_sd(ei, e, Eorig, s, d);
  float v = leaky(als[(size_t)s * H + hh] + ald[(size_t)d * H + hh]);
  float w = expf(v - segmax[(size_t)d * H + hh]);
  ew[i] = w;
  atomicAdd(&segsum[(size_t)d * H + hh], w);
}

// agg[dst, h, c] += h[src, h, c] * (ew[e,h] / segsum[dst,h])
// thread per (edge, channel); loop heads -> coalesced loads/atomics along c
__global__ void k_msg(const int* __restrict__ ei, const float* __restrict__ hfeat,
                      const float* __restrict__ ew, const float* __restrict__ segsum,
                      float* __restrict__ agg, int Etot, int Eorig, int H, int C) {
  int i = blockIdx.x * blockDim.x + threadIdx.x;
  if (i >= Etot * C) return;
  int e = i / C, c = i % C;
  int s, d; edge_sd(ei, e, Eorig, s, d);
  const float* hp = hfeat + (size_t)s * H * C + c;
  float*       ap = agg   + (size_t)d * H * C + c;
  for (int hh = 0; hh < H; ++hh) {
    float alpha = ew[(size_t)e * H + hh] / segsum[(size_t)d * H + hh];
    atomicAdd(ap + (size_t)hh * C, hp[(size_t)hh * C] * alpha);
  }
}

__global__ void k_bias_elu(float* __restrict__ a, const float* __restrict__ b,
                           int nNodes, int F) {
  int i = blockIdx.x * blockDim.x + threadIdx.x;
  if (i >= nNodes * F) return;
  float v = a[i] + b[i % F];
  a[i] = v > 0.f ? v : expm1f(v);
}

// ---------------- pooling + FC ----------------

__global__ void k_cnt(const int* __restrict__ batch, float* __restrict__ cnt, int n) {
  int i = blockIdx.x * blockDim.x + threadIdx.x;
  if (i < n) atomicAdd(&cnt[batch[i]], 1.0f);
}

__global__ void k_pool(const int* __restrict__ batch, const float* __restrict__ h,
                       float* __restrict__ pooled, int n, int C) {
  int i = blockIdx.x * blockDim.x + threadIdx.x;
  if (i >= n * C) return;
  int nn = i / C, c = i % C;
  atomicAdd(&pooled[(size_t)batch[nn] * C + c], h[i]);
}

__global__ void k_fc(const float* __restrict__ pooled, const float* __restrict__ cnt,
                     const float* __restrict__ W, const float* __restrict__ b,
                     float* __restrict__ out, int Gn, int C, int O) {
  int i = blockIdx.x * blockDim.x + threadIdx.x;
  if (i >= Gn * O) return;
  int g = i / O, o = i % O;
  float inv = 1.0f / fmaxf(cnt[g], 1.0f);
  float s = b[o];
  for (int c = 0; c < C; ++c) s += pooled[(size_t)g * C + c] * inv * W[(size_t)c * O + o];
  out[i] = s;
}

// ---------------- host orchestration ----------------

static void run_edges(hipStream_t stream, const int* ei, const float* hfeat,
                      const float* asrc, const float* adst, const float* bias,
                      float* als, float* ald, float* segmax, float* segsum,
                      float* ew, float* agg, int H, int C) {
  const int NH = NN * H, EH = ETOT * H, NF = NN * H * C, EC = ETOT * C;
  k_alpha   <<<CDIV(NH, 256), 256, 0, stream>>>(hfeat, asrc, adst, als, ald, NN, H, C);
  k_fill    <<<CDIV(NH, 256), 256, 0, stream>>>(segmax, -INFINITY, NH);
  k_fill    <<<CDIV(NH, 256), 256, 0, stream>>>(segsum, 0.0f, NH);
  k_edge_max<<<CDIV(EH, 256), 256, 0, stream>>>(ei, als, ald, segmax, ETOT, EE, H);
  k_edge_exp<<<CDIV(EH, 256), 256, 0, stream>>>(ei, als, ald, segmax, ew, segsum, ETOT, EE, H);
  k_fill    <<<CDIV(NF, 256), 256, 0, stream>>>(agg, 0.0f, NF);
  k_msg     <<<CDIV(EC, 256), 256, 0, stream>>>(ei, hfeat, ew, segsum, agg, ETOT, EE, H, C);
  k_bias_elu<<<CDIV(NF, 256), 256, 0, stream>>>(agg, bias, NN, H * C);
}

extern "C" void kernel_launch(void* const* d_in, const int* in_sizes, int n_in,
                              void* d_out, int out_size, void* d_ws, size_t ws_size,
                              hipStream_t stream) {
  const float* x    = (const float*)d_in[0];
  const int*   ei   = (const int*)d_in[1];
  const int*   batch= (const int*)d_in[2];
  const float* W1   = (const float*)d_in[3];
  const float* as1  = (const float*)d_in[4];
  const float* ad1  = (const float*)d_in[5];
  const float* b1   = (const float*)d_in[6];
  const float* W2   = (const float*)d_in[7];
  const float* as2  = (const float*)d_in[8];
  const float* ad2  = (const float*)d_in[9];
  const float* b2   = (const float*)d_in[10];
  const float* W3   = (const float*)d_in[11];
  const float* as3  = (const float*)d_in[12];
  const float* ad3  = (const float*)d_in[13];
  const float* b3   = (const float*)d_in[14];
  const float* Wfc  = (const float*)d_in[15];
  const float* bfc  = (const float*)d_in[16];
  float* out = (float*)d_out;

  // carve workspace (256B aligned chunks)
  char* p = (char*)d_ws;
  auto carve = [&](size_t bytes) -> void* {
    void* r = (void*)p;
    p += (bytes + 255) & ~(size_t)255;
    return r;
  };
  float*    hfeat  = (float*)   carve(sizeof(float)    * (size_t)NN * HCC);
  float*    agg    = (float*)   carve(sizeof(float)    * (size_t)NN * HCC);
  _Float16* acth   = (_Float16*)carve(sizeof(_Float16) * (size_t)NN * HCC);
  _Float16* w1t    = (_Float16*)carve(sizeof(_Float16) * (size_t)HCC * INC);
  _Float16* w2t    = (_Float16*)carve(sizeof(_Float16) * (size_t)HCC * HCC);
  _Float16* w3t    = (_Float16*)carve(sizeof(_Float16) * (size_t)HIDC * HCC);
  float*    als    = (float*)   carve(sizeof(float)    * (size_t)NN * HEADSC);
  float*    ald    = (float*)   carve(sizeof(float)    * (size_t)NN * HEADSC);
  float*    segmax = (float*)   carve(sizeof(float)    * (size_t)NN * HEADSC);
  float*    segsum = (float*)   carve(sizeof(float)    * (size_t)NN * HEADSC);
  float*    ew     = (float*)   carve(sizeof(float)    * (size_t)ETOT * HEADSC);
  float*    pooled = (float*)   carve(sizeof(float)    * (size_t)GC * HIDC);
  float*    cnt    = (float*)   carve(sizeof(float)    * (size_t)GC);

  dim3 blk(128);

  // ---- layer 1: in=256 -> heads*hid=512 ----
  k_f2h<<<CDIV(NN * INC, 256), 256, 0, stream>>>(x, acth, NN * INC);
  k_transpose_f2h<<<CDIV(INC * HCC, 256), 256, 0, stream>>>(W1, w1t, INC, HCC);
  k_gemm_wmma<<<dim3(NN / 16, HCC / 64), blk, 0, stream>>>(acth, w1t, hfeat, NN, INC, HCC);
  run_edges(stream, ei, hfeat, as1, ad1, b1, als, ald, segmax, segsum, ew, agg, HEADSC, HIDC);

  // ---- layer 2: 512 -> 512 ----
  k_f2h<<<CDIV(NN * HCC, 256), 256, 0, stream>>>(agg, acth, NN * HCC);
  k_transpose_f2h<<<CDIV(HCC * HCC, 256), 256, 0, stream>>>(W2, w2t, HCC, HCC);
  k_gemm_wmma<<<dim3(NN / 16, HCC / 64), blk, 0, stream>>>(acth, w2t, hfeat, NN, HCC, HCC);
  run_edges(stream, ei, hfeat, as2, ad2, b2, als, ald, segmax, segsum, ew, agg, HEADSC, HIDC);

  // ---- layer 3: 512 -> 64, heads=1 ----
  k_f2h<<<CDIV(NN * HCC, 256), 256, 0, stream>>>(agg, acth, NN * HCC);
  k_transpose_f2h<<<CDIV(HCC * HIDC, 256), 256, 0, stream>>>(W3, w3t, HCC, HIDC);
  k_gemm_wmma<<<dim3(NN / 16, HIDC / 64), blk, 0, stream>>>(acth, w3t, hfeat, NN, HCC, HIDC);
  run_edges(stream, ei, hfeat, as3, ad3, b3, als, ald, segmax, segsum, ew, agg, 1, HIDC);
  // agg[0 : NN*64) now holds elu(conv3 out)

  // ---- global mean pool + fc ----
  k_fill<<<CDIV(GC * HIDC, 256), 256, 0, stream>>>(pooled, 0.0f, GC * HIDC);
  k_fill<<<CDIV(GC, 256), 256, 0, stream>>>(cnt, 0.0f, GC);
  k_cnt <<<CDIV(NN, 256), 256, 0, stream>>>(batch, cnt, NN);
  k_pool<<<CDIV(NN * HIDC, 256), 256, 0, stream>>>(batch, agg, pooled, NN, HIDC);
  k_fc  <<<CDIV(GC * OUTC, 256), 256, 0, stream>>>(pooled, cnt, Wfc, bfc, out, GC, HIDC, OUTC);
}
